// GNNNet_15668040696494
// MI455X (gfx1250) — matrix operand order repaired and checked by reading, not compiled, not run on hardware
//
#include <hip/hip_runtime.h>
#include <math.h>

// ---------------------------------------------------------------------------
// GAT GNN forward for MI455X (gfx1250, wave32).
// GEMMs use v_wmma_f32_16x16x32_f16 (f16 inputs, f32 accumulate) with
// fragment-order f16 LDS tiles (convert-once staging via v_cvt_pk_rtz_f16_f32)
// + double buffering. Edge softmax uses L2-resident f32 atomics.
// ---------------------------------------------------------------------------

typedef __attribute__((ext_vector_type(16))) _Float16 v16h;
typedef __attribute__((ext_vector_type(8)))  float    v8f;

#define N_DRUG   20000
#define N_TARGET 20000
#define N_NODES  40000
#define EDGES    500000
#define DIM      128
#define NDm      (N_NODES * DIM)

// pack two f32 into packed f16x2 with a single v_cvt_pk_rtz_f16_f32
__device__ inline unsigned pack2h(float a, float b) {
  auto h = __builtin_amdgcn_cvt_pkrtz(a, b);   // half2, one instruction
  return __builtin_bit_cast(unsigned, h);
}

// ---------------------------------------------------------------------------
// Y[M x 128] = X[M x K] @ W[K x 128] (+ bias), fp32 in/out, f16 WMMA inside.
// Block: 256 threads = 8 waves. Block tile: 64 rows x 128 cols.
// Wave w owns cols [16w,16w+16) and 4 row-subtiles (reuses one B fragment 4x).
// LDS tiles are stored pre-swizzled in WMMA fragment order as packed f16, so
// a fragment build is one contiguous 32B LDS read per lane.
// ---------------------------------------------------------------------------
__global__ __launch_bounds__(256) void gemm_n128(
    const float* __restrict__ X, const float* __restrict__ Wt,
    const float* __restrict__ bias, float* __restrict__ Y, int M, int K) {
  // [buf][rb][lane][pair-slot] : per-lane 8 uints = 16 f16 = one A fragment
  __shared__ __align__(32) unsigned sAh[2][4][32][8];    // 2 x 4 KB
  // [buf][col][half][pair-slot] : per-lane 8 uints = one B fragment half
  __shared__ __align__(32) unsigned sBh[2][128][2][8];   // 2 x 8 KB

  const int tid  = threadIdx.x;
  const int wave = tid >> 5;
  const int lane = tid & 31;
  const int half = lane >> 4;
  const int l15  = lane & 15;
  const int row0 = blockIdx.x * 64;
  const int col0 = wave * 16;
  const int nk   = K >> 5;   // K is a multiple of 32

  v8f acc[4] = {v8f{}, v8f{}, v8f{}, v8f{}};

  // staging registers
  float4 ra0, ra1, rb0[2], rb1[2];
  const int a_r = (tid * 8) >> 5;       // A row handled by this thread (0..63)
  const int a_c = (tid * 8) & 31;       // A k-offset (multiple of 8)

  // A fragment-swizzle constants: this thread's 4 k-pairs land in the same
  // lane row, same half, at 4 consecutive pair slots.
  const int a_p0 = a_c >> 1;                      // 0,4,8,12
  const int a_hh = (a_p0 >> 2) & 1;               // lane-half
  const int a_v0 = 4 * (a_p0 >> 3);               // pair-slot base: 0 or 4
  const int a_ln = (a_hh << 4) | (a_r & 15);      // target lane row
  const int a_rb = a_r >> 4;                      // row subtile

  // ---- issue global loads for tile t into registers
  auto load_tile = [&](int t) {
    const int k0 = t << 5;
    ra0 = make_float4(0.f, 0.f, 0.f, 0.f);
    ra1 = ra0;
    if (row0 + a_r < M) {
      const float* src = X + (size_t)(row0 + a_r) * K + k0 + a_c;
      ra0 = *(const float4*)src;
      ra1 = *(const float4*)(src + 4);
    }
    #pragma unroll
    for (int j = 0; j < 2; ++j) {
      const int pc   = tid + j * 256;        // 0..511
      const int col4 = (pc & 31) * 4;        // column group
      const int kp   = pc >> 5;              // k-pair index 0..15
      rb0[j] = *(const float4*)(Wt + (size_t)(k0 + 2 * kp) * 128 + col4);
      rb1[j] = *(const float4*)(Wt + (size_t)(k0 + 2 * kp + 1) * 128 + col4);
    }
  };

  // ---- convert + store registers into fragment-order f16 LDS buffer
  auto store_tile = [&](int buf) {
    {
      uint4 pk;
      pk.x = pack2h(ra0.x, ra0.y);
      pk.y = pack2h(ra0.z, ra0.w);
      pk.z = pack2h(ra1.x, ra1.y);
      pk.w = pack2h(ra1.z, ra1.w);
      *(uint4*)&sAh[buf][a_rb][a_ln][a_v0] = pk;   // one ds_store_b128
    }
    #pragma unroll
    for (int j = 0; j < 2; ++j) {
      const int pc   = tid + j * 256;
      const int col4 = (pc & 31) * 4;
      const int kp   = pc >> 5;
      const int bh   = (kp >> 2) & 1, bs = (kp & 3) | ((kp >> 3) << 2);
      sBh[buf][col4 + 0][bh][bs] = pack2h(rb0[j].x, rb1[j].x);
      sBh[buf][col4 + 1][bh][bs] = pack2h(rb0[j].y, rb1[j].y);
      sBh[buf][col4 + 2][bh][bs] = pack2h(rb0[j].z, rb1[j].z);
      sBh[buf][col4 + 3][bh][bs] = pack2h(rb0[j].w, rb1[j].w);
    }
  };

  // ---- software pipeline: prologue
  load_tile(0);
  store_tile(0);
  __syncthreads();

  for (int t = 0; t < nk; ++t) {
    const int buf = t & 1;
    if (t + 1 < nk) load_tile(t + 1);   // global loads in flight during compute

    // B fragment: one contiguous 32B LDS read per lane
    v16h bfr = *(const v16h*)&sBh[buf][col0 + l15][half][0];
    #pragma unroll
    for (int rbi = 0; rbi < 4; ++rbi) {
      v16h afr = *(const v16h*)&sAh[buf][rbi][lane][0];
      acc[rbi] = __builtin_amdgcn_wmma_f32_16x16x32_f16(
          false, afr, false, bfr, (short)0, acc[rbi], false, false);
    }

    if (t + 1 < nk) {
      __syncthreads();                  // everyone done reading buf^1 (prev iter)
      store_tile(buf ^ 1);
      __syncthreads();                  // next tile visible to all waves
    }
  }

  // ---- C/D layout: VGPR v = row (v + 8*half), col = l15
  const float bcol = bias ? bias[col0 + l15] : 0.0f;
  #pragma unroll
  for (int rbi = 0; rbi < 4; ++rbi) {
    #pragma unroll
    for (int v = 0; v < 8; ++v) {
      const int r = row0 + rbi * 16 + v + 8 * half;
      if (r < M) Y[(size_t)r * 128 + col0 + l15] = acc[rbi][v] + bcol;
    }
  }
}

// ---------------------------------------------------------------------------
__global__ void fill_f32(float* p, float v, long long n) {
  long long i = blockIdx.x * (long long)blockDim.x + threadIdx.x;
  if (i < n) p[i] = v;
}

// per-edge dot: out[e] = edge_attr[e,:] . a[:]
__global__ void edge_dot(const float* __restrict__ attr,
                         const float* __restrict__ a, float* out, int ne) {
  int e = blockIdx.x * blockDim.x + threadIdx.x;
  if (e >= ne) return;
  const float4* row = (const float4*)(attr + (size_t)e * DIM);
  const float4* av  = (const float4*)a;
  float s = 0.f;
  #pragma unroll 8
  for (int j = 0; j < 32; ++j) {
    float4 h4 = row[j], a4 = av[j];
    s += h4.x * a4.x + h4.y * a4.y + h4.z * a4.z + h4.w * a4.w;
  }
  out[e] = s;
}

// per-node dots with a_src / a_dst
__global__ void node_dots(const float* __restrict__ hp,
                          const float* __restrict__ asv,
                          const float* __restrict__ adv,
                          float* s_src, float* s_dst, int n) {
  int i = blockIdx.x * blockDim.x + threadIdx.x;
  if (i >= n) return;
  const float4* row = (const float4*)(hp + (size_t)i * DIM);
  const float4* A   = (const float4*)asv;
  const float4* B   = (const float4*)adv;
  float ss = 0.f, sd = 0.f;
  #pragma unroll 8
  for (int j = 0; j < 32; ++j) {
    float4 h4 = row[j], a4 = A[j], b4 = B[j];
    ss += h4.x * a4.x + h4.y * a4.y + h4.z * a4.z + h4.w * a4.w;
    sd += h4.x * b4.x + h4.y * b4.y + h4.z * b4.z + h4.w * b4.w;
  }
  s_src[i] = ss;
  s_dst[i] = sd;
}

// ordered-int trick for float atomic max (values always finite here)
__device__ inline void atomicMaxFloat(float* addr, float val) {
  if (val >= 0.0f) atomicMax((int*)addr, __float_as_int(val));
  else             atomicMin((unsigned int*)addr, __float_as_uint(val));
}

// per-edge: logit = s_src[src]+s_dst[dst](+bias); leaky_relu; segment max on dst
__global__ void edge_logits(const int* __restrict__ ei,
                            const float* __restrict__ s_src,
                            const float* __restrict__ s_dst,
                            const float* __restrict__ bias,
                            float* elog, float* m, int ne) {
  int e = blockIdx.x * blockDim.x + threadIdx.x;
  if (e >= ne) return;
  int s = ei[e], d = ei[ne + e];
  float x = s_src[s] + s_dst[d];
  if (bias) x += bias[e];
  x = (x >= 0.f) ? x : 0.2f * x;   // leaky_relu(0.2)
  elog[e] = x;
  atomicMaxFloat(m + d, x);
}

// wave-per-edge: ex = exp(e - m[dst]); denom[dst]+=ex; num[dst,:] += ex*hp[src,:]
__global__ __launch_bounds__(256) void edge_scatter(
    const int* __restrict__ ei, const float* __restrict__ elog,
    const float* __restrict__ m, const float* __restrict__ hp,
    float* denom, float* num, int ne) {
  int wid  = (int)((blockIdx.x * (long long)blockDim.x + threadIdx.x) >> 5);
  int lane = threadIdx.x & 31;
  if (wid >= ne) return;
  int s = ei[wid], d = ei[ne + wid];
  float ex = __expf(elog[wid] - m[d]);   // m[d] finite: d has >=1 edge
  if (lane == 0) atomicAdd(denom + d, ex);
  float4 hv = *(const float4*)(hp + (size_t)s * DIM + lane * 4);
  float* np = num + (size_t)d * DIM + lane * 4;
  atomicAdd(np + 0, ex * hv.x);
  atomicAdd(np + 1, ex * hv.y);
  atomicAdd(np + 2, ex * hv.z);
  atomicAdd(np + 3, ex * hv.w);
}

// hnext += elu(num/denom) / 3
__global__ void finalize_accum(const float* __restrict__ num,
                               const float* __restrict__ denom,
                               float* hnext, int total) {
  int i = blockIdx.x * blockDim.x + threadIdx.x;
  if (i >= total) return;
  float x = num[i] / (denom[i >> 7] + 1e-16f);
  float e = (x > 0.f) ? x : (__expf(x) - 1.f);   // elu
  hnext[i] += e * (1.0f / 3.0f);
}

// ---------------------------------------------------------------------------
extern "C" void kernel_launch(void* const* d_in, const int* in_sizes, int n_in,
                              void* d_out, int out_size, void* d_ws,
                              size_t ws_size, hipStream_t stream) {
  const float* x_drug   = (const float*)d_in[0];
  const float* x_target = (const float*)d_in[1];
  const float* W_drug   = (const float*)d_in[2];
  const float* b_drug   = (const float*)d_in[3];
  const float* W_target = (const float*)d_in[4];
  const float* b_target = (const float*)d_in[5];
  const float* W_gat    = (const float*)d_in[6];   // [2,3,128,128]
  const float* a_src    = (const float*)d_in[7];   // [2,3,128]
  const float* a_dst    = (const float*)d_in[8];   // [2,3,128]
  const float* a_edge   = (const float*)d_in[9];   // [2,128]
  const float* e_attr   = (const float*)d_in[10];  // [E,128]
  const int*   edge_dd  = (const int*)d_in[11];    // [2,E]
  const int*   edge_dt  = (const int*)d_in[12];
  const int*   edge_tt  = (const int*)d_in[13];

  // workspace layout (floats): 4*ND + 4*N + 2*E  ~= 87 MB
  float* ws    = (float*)d_ws;
  float* hA    = ws;                 // N*D
  float* hB    = hA + NDm;           // N*D
  float* hp    = hB + NDm;           // N*D
  float* num   = hp + NDm;           // N*D
  float* ssrc  = num + NDm;          // N
  float* sdst  = ssrc + N_NODES;     // N
  float* mbuf  = sdst + N_NODES;     // N
  float* denom = mbuf + N_NODES;     // N
  float* elog  = denom + N_NODES;    // E
  float* biasb = elog + EDGES;       // E

  // ---- input projections (WMMA GEMMs)
  gemm_n128<<<(N_DRUG + 63) / 64, 256, 0, stream>>>(
      x_drug, W_drug, b_drug, hA, N_DRUG, 2048);
  gemm_n128<<<(N_TARGET + 63) / 64, 256, 0, stream>>>(
      x_target, W_target, b_target, hA + (size_t)N_DRUG * DIM, N_TARGET, 1280);

  const int* edges[3] = {edge_dd, edge_dt, edge_tt};
  float* hin = hA;
  for (int l = 0; l < 2; ++l) {
    float* hout = (l == 0) ? hB : (float*)d_out;
    fill_f32<<<(NDm + 255) / 256, 256, 0, stream>>>(hout, 0.0f, NDm);
    edge_dot<<<(EDGES + 255) / 256, 256, 0, stream>>>(
        e_attr, a_edge + l * DIM, biasb, EDGES);

    for (int t = 0; t < 3; ++t) {
      // hp = hin @ W_gat[l][t]   (WMMA GEMM, K=128)
      gemm_n128<<<(N_NODES + 63) / 64, 256, 0, stream>>>(
          hin, W_gat + (size_t)(l * 3 + t) * DIM * DIM, nullptr, hp,
          N_NODES, DIM);
      node_dots<<<(N_NODES + 255) / 256, 256, 0, stream>>>(
          hp, a_src + (l * 3 + t) * DIM, a_dst + (l * 3 + t) * DIM,
          ssrc, sdst, N_NODES);
      fill_f32<<<(NDm + 255) / 256, 256, 0, stream>>>(num, 0.0f, NDm);
      fill_f32<<<(N_NODES + 255) / 256, 256, 0, stream>>>(denom, 0.0f, N_NODES);
      fill_f32<<<(N_NODES + 255) / 256, 256, 0, stream>>>(mbuf, -INFINITY, N_NODES);
      edge_logits<<<(EDGES + 255) / 256, 256, 0, stream>>>(
          edges[t], ssrc, sdst, (t == 0) ? biasb : nullptr, elog, mbuf, EDGES);
      edge_scatter<<<EDGES / 8, 256, 0, stream>>>(
          edges[t], elog, mbuf, hp, denom, num, EDGES);
      finalize_accum<<<(NDm + 255) / 256, 256, 0, stream>>>(
          num, denom, hout, NDm);
    }
    hin = hout;
  }
  // d_out == final h = concat(h_drug, h_target), written in place by layer 1.
}
